// LIFNet_5574867550399
// MI455X (gfx1250) — matrix operand order repaired and checked by reading, not compiled, not run on hardware
//
#include <hip/hip_runtime.h>
#include <stdint.h>
#include <math.h>

#define N_NEUR 4096
#define T_STEPS 2000
#define DT_F 0.001f

// plain clang ext vectors (union-safe, no HIP struct ctors)
typedef __attribute__((ext_vector_type(16))) __bf16   v16bf;
typedef __attribute__((ext_vector_type(8)))  float    v8f;
typedef __attribute__((ext_vector_type(16))) float    v16f;
typedef __attribute__((ext_vector_type(4)))  uint32_t u32x4;
typedef __attribute__((ext_vector_type(2)))  uint32_t u32x2;
typedef __attribute__((ext_vector_type(4)))  float    f32x4;

// ---- workspace layout ----
// float-indexed regions (ws as float*):
static constexpr size_t OFF_V   = 0;          // voltage state      [4096]
static constexpr size_t OFF_C   = 4096;       // synaptic current   [4096]
static constexpr size_t OFF_TMR = 8192;       // refractory timer   [4096]
static constexpr size_t OFF_P00 = 12288;      // exp(-dt/tau_m)
static constexpr size_t OFF_P01 = 16384;      // current->voltage propagator
static constexpr size_t OFF_P02 = 20480;      // stimulus propagator
static constexpr size_t OFF_P11 = 24576;      // exp(-dt/tau_s)
static constexpr size_t OFF_ITS = 28672;      // 1/tau_s
// byte offsets:
static constexpr size_t SBUF_BYTE = 131072;   // spike bf16, double buffered: 2 x 4096 x 2B
static constexpr size_t ZERO_BYTE = 147456;   // 4096 bf16 zeros (A-operand zero page)
static constexpr size_t WBF_BYTE  = 262144;   // W as bf16: 4096*4096*2 = 32MB

__global__ void lif_init(const float* __restrict__ v0, const float* __restrict__ c0,
                         const float* __restrict__ tau_m, const float* __restrict__ tau_s,
                         float* __restrict__ ws) {
    int n = blockIdx.x * blockDim.x + threadIdx.x;
    if (n >= N_NEUR) return;
    float tm = tau_m[n], tsv = tau_s[n];
    float em = expf(-DT_F / tm);
    float es = expf(-DT_F / tsv);
    float diff = tsv - tm;
    float mul = (diff == 0.0f) ? 0.0f : (tsv * tm) / diff;
    ws[OFF_V + n]   = v0[n];
    ws[OFF_C + n]   = c0[n];
    ws[OFF_TMR + n] = 0.0f;
    ws[OFF_P00 + n] = em;
    ws[OFF_P01 + n] = mul * (es - em);
    ws[OFF_P02 + n] = tm * (1.0f - em);
    ws[OFF_P11 + n] = es;
    ws[OFF_ITS + n] = 1.0f / tsv;
    uint16_t* sb = (uint16_t*)((char*)ws + SBUF_BYTE);
    sb[n] = 0; sb[N_NEUR + n] = 0;
    uint16_t* zp = (uint16_t*)((char*)ws + ZERO_BYTE);
    zp[n] = 0;
}

__device__ __forceinline__ uint16_t f2bf_bits(float f) {
    return __builtin_bit_cast(uint16_t, (__bf16)f);
}

// W (fp32, row-major) -> bf16, 4 elements per thread
__global__ void lif_wcvt(const float* __restrict__ W, uint16_t* __restrict__ Wb) {
    size_t i = ((size_t)blockIdx.x * blockDim.x + threadIdx.x) * 4;
    f32x4 f = *(const f32x4*)(W + i);
    u32x2 o;
    o[0] = (uint32_t)f2bf_bits(f[0]) | ((uint32_t)f2bf_bits(f[1]) << 16);
    o[1] = (uint32_t)f2bf_bits(f[2]) | ((uint32_t)f2bf_bits(f[3]) << 16);
    *(u32x2*)(Wb + i) = o;
}

__device__ __forceinline__ v8f wmma_bf16(v16bf a, v16bf b, v8f c) {
    // (neg_a, A, neg_b, B, c_mod, C, reuse_a, reuse_b)
    return __builtin_amdgcn_wmma_f32_16x16x32_bf16(false, a, false, b, (short)0, c,
                                                   false, false);
}

// Launch L: [MATVEC] finish step t-1: c += (W@s_{t-1} + jext@ext[t-1])/tau_s, write C[t-1]
//           [UPDATE] pointwise step t: membrane update, spikes, write V[t], S[t], s_buf[t&1]
// One wave of 32 lanes owns 16 neurons (lane&15 selects neuron / B column; lane>>4 = K-half).
template <bool MATVEC, bool UPDATE, bool WSBF>
__global__ __launch_bounds__(256) void lif_step(
    float* __restrict__ ws, const uint16_t* __restrict__ wbf, const float* __restrict__ W,
    const float* __restrict__ stim, const float* __restrict__ ext,
    const float* __restrict__ jext, const float* __restrict__ tau_ref,
    const float* __restrict__ thr, const float* __restrict__ rst,
    float* __restrict__ Vout, float* __restrict__ Cout, float* __restrict__ Sout, int t)
{
    const int tid  = blockIdx.x * 256 + threadIdx.x;
    const int lane = threadIdx.x & 31;
    const int col  = lane & 15;   // output neuron within wave; B-matrix column N
    const int kh   = lane >> 4;   // 0: K 0-15, 1: K 16-31 within a 32-chunk
    const int n    = (tid >> 5) * 16 + col;

    float* vst = ws + OFF_V;
    float* cst = ws + OFF_C;
    float* tmr = ws + OFF_TMR;
    const float* p00 = ws + OFF_P00;
    const float* p01 = ws + OFF_P01;
    const float* p02 = ws + OFF_P02;
    const float* p11 = ws + OFF_P11;
    const float* its = ws + OFF_ITS;
    uint16_t* sb       = (uint16_t*)((char*)ws + SBUF_BYTE);
    const uint16_t* zp = (const uint16_t*)((char*)ws + ZERO_BYTE);

    float cc = 0.0f;

    if constexpr (MATVEC) {
        const int tp = t - 1;
        const uint16_t* sp = sb + (size_t)(tp & 1) * N_NEUR;
        // A-matrix: only row M=0 (lanes with col==0) carries the spike vector;
        // every other lane reads a zero page -> pointer select once, EXEC stays all-1s.
        const uint16_t* sr = ((col == 0) ? sp : zp) + kh * 16;

        v8f acc0 = {}; v8f acc1 = {};
        union U { u32x4 u[2]; v16bf v; };

        if constexpr (WSBF) {
            // B tile: lane holds W[n][k0 + kh*16 .. +16) -- contiguous bf16 from row-major W
            const uint16_t* wr = wbf + (size_t)n * N_NEUR + kh * 16;
            for (int k0 = 0; k0 < N_NEUR; k0 += 64) {
                U a0, a1, b0, b1;
                b0.u[0] = *(const u32x4*)(wr + k0);
                b0.u[1] = *(const u32x4*)(wr + k0 + 8);
                b1.u[0] = *(const u32x4*)(wr + k0 + 32);
                b1.u[1] = *(const u32x4*)(wr + k0 + 40);
                a0.u[0] = *(const u32x4*)(sr + k0);        // K 0-7   (this kh)
                a0.u[1] = *(const u32x4*)(sr + k0 + 8);    // K 8-15  (this kh)
                a1.u[0] = *(const u32x4*)(sr + k0 + 32);
                a1.u[1] = *(const u32x4*)(sr + k0 + 40);
                acc0 = wmma_bf16(a0.v, b0.v, acc0);
                acc1 = wmma_bf16(a1.v, b1.v, acc1);
            }
        } else {
            // fallback: fp32 W from HBM/L2, convert to bf16 in-register
            const float* wr = W + (size_t)n * N_NEUR + kh * 16;
            for (int k0 = 0; k0 < N_NEUR; k0 += 64) {
                union F { f32x4 f[4]; v16f v; } w0, w1;
                w0.f[0] = *(const f32x4*)(wr + k0);
                w0.f[1] = *(const f32x4*)(wr + k0 + 4);
                w0.f[2] = *(const f32x4*)(wr + k0 + 8);
                w0.f[3] = *(const f32x4*)(wr + k0 + 12);
                w1.f[0] = *(const f32x4*)(wr + k0 + 32);
                w1.f[1] = *(const f32x4*)(wr + k0 + 36);
                w1.f[2] = *(const f32x4*)(wr + k0 + 40);
                w1.f[3] = *(const f32x4*)(wr + k0 + 44);
                v16bf b0v, b1v;
#pragma unroll
                for (int i = 0; i < 16; ++i) {
                    b0v[i] = (__bf16)w0.v[i];
                    b1v[i] = (__bf16)w1.v[i];
                }
                U a0, a1;
                a0.u[0] = *(const u32x4*)(sr + k0);
                a0.u[1] = *(const u32x4*)(sr + k0 + 8);
                a1.u[0] = *(const u32x4*)(sr + k0 + 32);
                a1.u[1] = *(const u32x4*)(sr + k0 + 40);
                acc0 = wmma_bf16(a0.v, b0v, acc0);
                acc1 = wmma_bf16(a1.v, b1v, acc1);
            }
        }

        if (lane < 16) {
            // D row M=0 lives in element 0 (VGPR0) of lanes 0-15, N = lane
            float m  = acc0[0] + acc1[0];
            float e0 = ext[((size_t)tp * 2 + 0) * N_NEUR + n];
            float e1 = ext[((size_t)tp * 2 + 1) * N_NEUR + n];
            cc = cst[n] + (m + jext[0] * e0 + jext[1] * e1) * its[n];
            Cout[(size_t)tp * N_NEUR + n] = cc;
        }
    } else {
        if (lane < 16) cc = cst[n];
    }

    if (lane < 16) {
        if constexpr (UPDATE) {
            float v  = vst[n];
            float tr = tmr[n];
            bool is_ref = fabsf(tr) >= 5e-7f;  // round(timer, 6) != 0
            float v_upd = v * p00[n] + cc * p01[n] + stim[(size_t)t * N_NEUR + n] * p02[n];
            float cn = cc * p11[n];
            v = is_ref ? rst[n] : v_upd;
            bool spk = v >= thr[n];
            v  = spk ? rst[n] : v;
            tr = spk ? tau_ref[n] : tr;
            Vout[(size_t)t * N_NEUR + n] = v;
            Sout[(size_t)t * N_NEUR + n] = spk ? 1.0f : 0.0f;
            sb[(size_t)(t & 1) * N_NEUR + n] = spk ? (uint16_t)0x3F80 : (uint16_t)0;  // bf16 1.0/0.0
            if (is_ref) tr -= DT_F;
            vst[n] = v;
            cst[n] = cn;
            tmr[n] = tr;
        } else {
            cst[n] = cc;  // final launch: commit post-matvec state
        }
    }
}

extern "C" void kernel_launch(void* const* d_in, const int* in_sizes, int n_in,
                              void* d_out, int out_size, void* d_ws, size_t ws_size,
                              hipStream_t stream) {
    (void)in_sizes; (void)n_in; (void)out_size;
    const float* v0      = (const float*)d_in[0];
    const float* c0      = (const float*)d_in[1];
    const float* stim    = (const float*)d_in[2];
    const float* ext     = (const float*)d_in[3];
    const float* W       = (const float*)d_in[4];
    const float* jext    = (const float*)d_in[5];
    const float* tau_ref = (const float*)d_in[8];
    const float* thr     = (const float*)d_in[9];
    const float* rst     = (const float*)d_in[10];

    float* out  = (float*)d_out;
    float* Vout = out;
    float* Cout = out + (size_t)T_STEPS * N_NEUR;
    float* Sout = out + 2 * (size_t)T_STEPS * N_NEUR;

    float*    ws  = (float*)d_ws;
    uint16_t* wbf = (uint16_t*)((char*)d_ws + WBF_BYTE);
    const bool usebf = ws_size >= WBF_BYTE + (size_t)N_NEUR * N_NEUR * 2;

    const dim3 blk(256);
    const dim3 grid(32);  // 32 blocks * 8 waves = 256 waves * 16 neurons = 4096

    lif_init<<<dim3(16), blk, 0, stream>>>(v0, c0, (const float*)d_in[6],
                                           (const float*)d_in[7], ws);
    if (usebf)
        lif_wcvt<<<dim3((N_NEUR * N_NEUR) / (4 * 256)), blk, 0, stream>>>(W, wbf);

    // L=0: update only (no prior spikes)
    lif_step<false, true, true><<<grid, blk, 0, stream>>>(
        ws, wbf, W, stim, ext, jext, tau_ref, thr, rst, Vout, Cout, Sout, 0);

    if (usebf) {
        for (int L = 1; L < T_STEPS; ++L)
            lif_step<true, true, true><<<grid, blk, 0, stream>>>(
                ws, wbf, W, stim, ext, jext, tau_ref, thr, rst, Vout, Cout, Sout, L);
        lif_step<true, false, true><<<grid, blk, 0, stream>>>(
            ws, wbf, W, stim, ext, jext, tau_ref, thr, rst, Vout, Cout, Sout, T_STEPS);
    } else {
        for (int L = 1; L < T_STEPS; ++L)
            lif_step<true, true, false><<<grid, blk, 0, stream>>>(
                ws, wbf, W, stim, ext, jext, tau_ref, thr, rst, Vout, Cout, Sout, L);
        lif_step<true, false, false><<<grid, blk, 0, stream>>>(
            ws, wbf, W, stim, ext, jext, tau_ref, thr, rst, Vout, Cout, Sout, T_STEPS);
    }
}
